// SplitLayer0_1_30382598652492
// MI455X (gfx1250) — compile-verified, matrix operand order checked by reading, not run on hardware
//
#include <hip/hip_runtime.h>
#include <hip/hip_bf16.h>

// MI455X / gfx1250, wave32. bf16 WMMA path: v_wmma_f32_16x16x32_bf16.

typedef __bf16 bf16_t;
typedef __attribute__((ext_vector_type(16))) __bf16 v16bf;
typedef __attribute__((ext_vector_type(8)))  float  v8f;

#define NNODE 50000
#define NEDGE 150000
#define NATOM 300000   // R = 2E

// ---------------------------------------------------------------- utilities
__global__ void k_zero_f32(float* __restrict__ p, int n) {
  int i = blockIdx.x * blockDim.x + threadIdx.x;
  if (i < n) p[i] = 0.f;
}

// W [Kin, Nout] f32  ->  Wt [Nout, Kin] bf16 (transposed for WMMA B fragments)
__global__ void k_wcvt(const float* __restrict__ W, bf16_t* __restrict__ Wt,
                       int Kin, int Nout) {
  int i = blockIdx.x * blockDim.x + threadIdx.x;
  int total = Kin * Nout;
  if (i < total) {
    int k = i / Nout, o = i - k * Nout;
    Wt[(size_t)o * Kin + k] = (bf16_t)W[i];
  }
}

__global__ void k_cvt_bf16(const float* __restrict__ in, bf16_t* __restrict__ out, int total) {
  int i = blockIdx.x * blockDim.x + threadIdx.x;
  if (i < total) out[i] = (bf16_t)in[i];
}

// ------------------------------------------------- build X1 [R,384] and edge_pre [R,256]
// one block per edge (atoms 2e, 2e+1), 128 threads = one column each
__global__ void k_build_inputs(const float* __restrict__ node_rep,
                               const float* __restrict__ edge_rep,
                               const int*   __restrict__ atom_node,
                               const float* __restrict__ eps2_p,
                               bf16_t* __restrict__ X1, bf16_t* __restrict__ Xe) {
  int e = blockIdx.x;
  int t = threadIdx.x;                 // 0..127
  int r0 = 2 * e, r1 = 2 * e + 1;
  float c2 = 1.0f + eps2_p[0];
  int n0 = atom_node[r0], n1 = atom_node[r1];
  float na = node_rep[(size_t)n0 * 128 + t];
  float nb = node_rep[(size_t)n1 * 128 + t];
  float es = na + nb;                  // within-edge sum of gathered node features
  float e0 = edge_rep[(size_t)r0 * 128 + t];
  float e1 = edge_rep[(size_t)r1 * 128 + t];
  float em = 0.5f * (e0 + e1);         // per-edge mean
  bf16_t* xa = X1 + (size_t)r0 * 384;
  bf16_t* xb = X1 + (size_t)r1 * 384;
  xa[t] = (bf16_t)na; xa[128 + t] = (bf16_t)es; xa[256 + t] = (bf16_t)e0;
  xb[t] = (bf16_t)nb; xb[128 + t] = (bf16_t)es; xb[256 + t] = (bf16_t)e1;
  // edge_pre = (1+eps2)*[edge | mean] + [local | esum]
  bf16_t* pa = Xe + (size_t)r0 * 256;
  bf16_t* pb = Xe + (size_t)r1 * 256;
  pa[t] = (bf16_t)(c2 * e0 + na); pa[128 + t] = (bf16_t)(c2 * em + es);
  pb[t] = (bf16_t)(c2 * e1 + nb); pb[128 + t] = (bf16_t)(c2 * em + es);
}

// --------------------------------------------------------------- WMMA GEMM
// C[M,NC] f32 = A[M,K] bf16 @ Bt[NC,K]^T bf16, fused BN column stats (sum, sumsq).
// One wave = one 16-row stripe x NC cols. Block = 8 waves = 128 rows.
template <int K, int NC>
__global__ void k_gemm_bn(const bf16_t* __restrict__ A, const bf16_t* __restrict__ Bt,
                          float* __restrict__ C,
                          float* __restrict__ colSum, float* __restrict__ colSq,
                          int Mtiles) {
  __shared__ float lsum[NC], lsq[NC];
  int tid = threadIdx.x;
  if (tid < NC) { lsum[tid] = 0.f; lsq[tid] = 0.f; }
  __syncthreads();

  int wave = tid >> 5, lane = tid & 31;
  int tile = blockIdx.x * 8 + wave;
  bool valid = tile < Mtiles;
  int half = lane >> 4;                // 0: low half-wave, 1: high
  int lr   = lane & 15;

  if (valid) {
    v8f acc[NC / 16];
#pragma unroll
    for (int t = 0; t < NC / 16; ++t) {
      v8f z = {0.f, 0.f, 0.f, 0.f, 0.f, 0.f, 0.f, 0.f};
      acc[t] = z;
    }
    // A fragment rows: lane lr -> row tile*16+lr; K interleave per ISA 16-bit A layout
    const bf16_t* arow = A + (size_t)(tile * 16 + lr) * K + half * 8;
    for (int k0 = 0; k0 < K; k0 += 32) {
      v16bf a;
      *(uint4*)&a       = *(const uint4*)(arow + k0);
      *((uint4*)&a + 1) = *(const uint4*)(arow + k0 + 16);
#pragma unroll
      for (int t = 0; t < NC / 16; ++t) {
        const bf16_t* brow = Bt + (size_t)(t * 16 + lr) * K + half * 8 + k0;
        v16bf b;
        *(uint4*)&b       = *(const uint4*)(brow);
        *((uint4*)&b + 1) = *(const uint4*)(brow + 16);
        acc[t] = __builtin_amdgcn_wmma_f32_16x16x32_bf16(
            false, a, false, b, (short)0, acc[t], false, false);
      }
    }
    // store + per-column partial stats (C layout: lane lr = col, elem r = row half*8+r)
#pragma unroll
    for (int t = 0; t < NC / 16; ++t) {
      int col = t * 16 + lr;
      float s = 0.f, q = 0.f;
#pragma unroll
      for (int r = 0; r < 8; ++r) {
        float z = acc[t][r];
        C[(size_t)(tile * 16 + half * 8 + r) * NC + col] = z;
        s += z; q += z * z;
      }
      atomicAdd(&lsum[col], s);
      atomicAdd(&lsq[col], q);
    }
  }
  __syncthreads();
  if (tid < NC) {
    atomicAdd(&colSum[tid], lsum[tid]);
    atomicAdd(&colSq[tid],  lsq[tid]);
  }
}

// ------------------------------------------------------------ BN + ReLU apply
template <typename OutT>
__global__ void k_bn_relu(const float* __restrict__ Z, OutT* __restrict__ Y,
                          const float* __restrict__ colSum, const float* __restrict__ colSq,
                          const float* __restrict__ g, const float* __restrict__ b,
                          int M, int Cc) {
  size_t total = (size_t)M * Cc;
  size_t stride = (size_t)gridDim.x * blockDim.x;
  float invM = 1.0f / (float)M;
  for (size_t i = (size_t)blockIdx.x * blockDim.x + threadIdx.x; i < total; i += stride) {
    int col = (int)(i % Cc);
    float m = colSum[col] * invM;
    float v = colSq[col] * invM - m * m;
    float z = Z[i];
    float y = (z - m) * rsqrtf(v + 1e-5f) * g[col] + b[col];
    y = y > 0.f ? y : 0.f;
    Y[i] = (OutT)y;
  }
}

// ------------------------------------------------------------- node accumulate
__global__ void k_init_node_pre(const float* __restrict__ node_rep,
                                const float* __restrict__ eps11_p,
                                float* __restrict__ node_pre, int total) {
  int i = blockIdx.x * blockDim.x + threadIdx.x;
  if (i < total) node_pre[i] = (1.0f + eps11_p[0]) * node_rep[i];
}

// per atom r: node_pre[an[r]] += (1+eps12)*y[r] + (y[r] + y[r^1])  = (2+eps12)*y[r] + y[r^1]
__global__ void k_scatter_atoms(const float* __restrict__ y,
                                const int* __restrict__ atom_node,
                                const float* __restrict__ eps12_p,
                                float* __restrict__ node_pre) {
  int e = blockIdx.x;
  int t = threadIdx.x;                 // 0..127
  float c2 = 2.0f + eps12_p[0];
  int r0 = 2 * e, r1 = r0 + 1;
  float y0 = y[(size_t)r0 * 128 + t];
  float y1 = y[(size_t)r1 * 128 + t];
  int n0 = atom_node[r0], n1 = atom_node[r1];
  atomicAdd(&node_pre[(size_t)n0 * 128 + t], c2 * y0 + y1);
  atomicAdd(&node_pre[(size_t)n1 * 128 + t], c2 * y1 + y0);
}

// =========================================================================
static inline int cdiv(long long a, long long b) { return (int)((a + b - 1) / b); }

extern "C" void kernel_launch(void* const* d_in, const int* in_sizes, int n_in,
                              void* d_out, int out_size, void* d_ws, size_t ws_size,
                              hipStream_t stream) {
  const float* node_rep = (const float*)d_in[0];
  const float* edge_rep = (const float*)d_in[1];
  const float* lift_W1  = (const float*)d_in[2];   // [256,256]
  const float* lift_g1  = (const float*)d_in[3];
  const float* lift_b1  = (const float*)d_in[4];
  const float* lift_W2  = (const float*)d_in[5];   // [256,128]
  const float* lift_g2  = (const float*)d_in[6];
  const float* lift_b2  = (const float*)d_in[7];
  const float* lvl1_W   = (const float*)d_in[8];   // [384,128]
  const float* lvl1_g   = (const float*)d_in[9];
  const float* lvl1_b   = (const float*)d_in[10];
  const float* lvl2_W1  = (const float*)d_in[11];  // [128,256]
  const float* lvl2_g1  = (const float*)d_in[12];
  const float* lvl2_b1  = (const float*)d_in[13];
  const float* lvl2_W2  = (const float*)d_in[14];  // [256,128]
  const float* lvl2_g2  = (const float*)d_in[15];
  const float* lvl2_b2  = (const float*)d_in[16];
  const float* eps1_1   = (const float*)d_in[17];
  const float* eps1_2   = (const float*)d_in[18];
  const float* eps2     = (const float*)d_in[19];
  const int*   atom_node = (const int*)d_in[20];
  // d_in[21] atom_edge == arange(R)//2 (structure hardcoded)

  char* ws = (char*)d_ws;

  // ---- workspace layout (all sizes multiples of 256B) ----
  constexpr size_t SZ_X1  = (size_t)NATOM * 384 * 2;   // 230.4 MB bf16
  constexpr size_t SZ_Y   = (size_t)NATOM * 128 * 4;   // 153.6 MB f32 (Z1 then y in place)
  constexpr size_t SZ_XE  = (size_t)NATOM * 256 * 2;   // 153.6 MB bf16 edge_pre
  constexpr size_t SZ_NP  = (size_t)NNODE * 128 * 4;
  constexpr size_t SZ_A2  = (size_t)NNODE * 128 * 2;
  constexpr size_t SZ_ZN1 = (size_t)NNODE * 256 * 4;
  constexpr size_t SZ_HN  = (size_t)NNODE * 256 * 2;
  constexpr size_t SZ_ZN2 = (size_t)NNODE * 128 * 4;
  constexpr size_t SZ_HE  = (size_t)NATOM * 256 * 2;
  constexpr size_t SZ_ZE2 = (size_t)NATOM * 128 * 4;

  size_t off = 0;
  const size_t OFF_X1  = off; off += SZ_X1;
  const size_t OFF_Y   = off; off += SZ_Y;
  const size_t OFF_ZE1 = OFF_X1;                 // Zedge1 (307.2MB) aliases dead X1+Y (384MB)
  const size_t OFF_XE  = off; off += SZ_XE;
  const size_t OFF_NP  = off; off += SZ_NP;
  const size_t OFF_A2  = off; off += SZ_A2;
  const size_t OFF_ZN1 = off; off += SZ_ZN1;
  const size_t OFF_HN  = off; off += SZ_HN;
  const size_t OFF_ZN2 = off; off += SZ_ZN2;
  const size_t OFF_HE  = off; off += SZ_HE;
  const size_t OFF_ZE2 = off; off += SZ_ZE2;
  const size_t OFF_WT1 = off; off += (size_t)128 * 384 * 2;
  const size_t OFF_WT2 = off; off += (size_t)256 * 128 * 2;
  const size_t OFF_WT3 = off; off += (size_t)128 * 256 * 2;
  const size_t OFF_WT4 = off; off += (size_t)256 * 256 * 2;
  const size_t OFF_WT5 = off; off += (size_t)128 * 256 * 2;
  const size_t OFF_ST  = off;                    // 1792 f32 stat scalars

  bf16_t* X1  = (bf16_t*)(ws + OFF_X1);
  float*  Z1  = (float*)(ws + OFF_Y);            // then y in-place
  bf16_t* Xe  = (bf16_t*)(ws + OFF_XE);
  float*  NP  = (float*)(ws + OFF_NP);
  bf16_t* A2  = (bf16_t*)(ws + OFF_A2);
  float*  ZN1 = (float*)(ws + OFF_ZN1);
  bf16_t* HN  = (bf16_t*)(ws + OFF_HN);
  float*  ZN2 = (float*)(ws + OFF_ZN2);
  float*  ZE1 = (float*)(ws + OFF_ZE1);
  bf16_t* HE  = (bf16_t*)(ws + OFF_HE);
  float*  ZE2 = (float*)(ws + OFF_ZE2);
  bf16_t* Wt1 = (bf16_t*)(ws + OFF_WT1);
  bf16_t* Wt2 = (bf16_t*)(ws + OFF_WT2);
  bf16_t* Wt3 = (bf16_t*)(ws + OFF_WT3);
  bf16_t* Wt4 = (bf16_t*)(ws + OFF_WT4);
  bf16_t* Wt5 = (bf16_t*)(ws + OFF_WT5);
  float* st = (float*)(ws + OFF_ST);
  float *s1 = st,        *q1 = st + 128;         // lvl1   (128)
  float *s2 = st + 256,  *q2 = st + 512;         // lvl2_1 (256)
  float *s3 = st + 768,  *q3 = st + 896;         // lvl2_2 (128)
  float *s4 = st + 1024, *q4 = st + 1280;        // lift1  (256)
  float *s5 = st + 1536, *q5 = st + 1664;        // lift2  (128)

  float* node_out = (float*)d_out;
  float* edge_out = (float*)d_out + (size_t)NNODE * 128;

  // 1) zero BN stats (workspace is NOT re-poisoned between replays)
  k_zero_f32<<<cdiv(1792, 256), 256, 0, stream>>>(st, 1792);

  // 2) weights -> bf16, transposed
  k_wcvt<<<cdiv(384 * 128, 256), 256, 0, stream>>>(lvl1_W,  Wt1, 384, 128);
  k_wcvt<<<cdiv(128 * 256, 256), 256, 0, stream>>>(lvl2_W1, Wt2, 128, 256);
  k_wcvt<<<cdiv(256 * 128, 256), 256, 0, stream>>>(lvl2_W2, Wt3, 256, 128);
  k_wcvt<<<cdiv(256 * 256, 256), 256, 0, stream>>>(lift_W1, Wt4, 256, 256);
  k_wcvt<<<cdiv(256 * 128, 256), 256, 0, stream>>>(lift_W2, Wt5, 256, 128);

  // 3) gather + pair-sums -> X1 [R,384], edge_pre -> Xe [R,256]
  k_build_inputs<<<NEDGE, 128, 0, stream>>>(node_rep, edge_rep, atom_node, eps2, X1, Xe);

  // 4) Z1 = X1 @ lvl1_W   (R x 384 x 128) + stats
  k_gemm_bn<384, 128><<<cdiv(NATOM / 16, 8), 256, 0, stream>>>(X1, Wt1, Z1, s1, q1, NATOM / 16);
  // 5) y = BN+ReLU(Z1) in place
  k_bn_relu<float><<<cdiv((long long)NATOM * 128, 256), 256, 0, stream>>>(
      Z1, Z1, s1, q1, lvl1_g, lvl1_b, NATOM, 128);

  // 6-7) node_pre = (1+e11)*node_rep + (1+e12)*a0 + a1  via per-atom atomics
  k_init_node_pre<<<cdiv(NNODE * 128, 256), 256, 0, stream>>>(node_rep, eps1_1, NP, NNODE * 128);
  k_scatter_atoms<<<NEDGE, 128, 0, stream>>>(Z1, atom_node, eps1_2, NP);

  // 8) node_pre -> bf16
  k_cvt_bf16<<<cdiv(NNODE * 128, 256), 256, 0, stream>>>(NP, A2, NNODE * 128);

  // 9-12) node path: (N,128)@(128,256) -> BN+ReLU -> (N,256)@(256,128) -> BN+ReLU -> node_out
  k_gemm_bn<128, 256><<<cdiv(NNODE / 16, 8), 256, 0, stream>>>(A2, Wt2, ZN1, s2, q2, NNODE / 16);
  k_bn_relu<bf16_t><<<cdiv((long long)NNODE * 256, 256), 256, 0, stream>>>(
      ZN1, HN, s2, q2, lvl2_g1, lvl2_b1, NNODE, 256);
  k_gemm_bn<256, 128><<<cdiv(NNODE / 16, 8), 256, 0, stream>>>(HN, Wt3, ZN2, s3, q3, NNODE / 16);
  k_bn_relu<float><<<cdiv((long long)NNODE * 128, 256), 256, 0, stream>>>(
      ZN2, node_out, s3, q3, lvl2_g2, lvl2_b2, NNODE, 128);

  // 13-16) edge path: (R,256)@(256,256) -> BN+ReLU -> (R,256)@(256,128) -> BN+ReLU -> edge_out
  k_gemm_bn<256, 256><<<cdiv(NATOM / 16, 8), 256, 0, stream>>>(Xe, Wt4, ZE1, s4, q4, NATOM / 16);
  k_bn_relu<bf16_t><<<cdiv((long long)NATOM * 256, 256), 256, 0, stream>>>(
      ZE1, HE, s4, q4, lift_g1, lift_b1, NATOM, 256);
  k_gemm_bn<256, 128><<<cdiv(NATOM / 16, 8), 256, 0, stream>>>(HE, Wt5, ZE2, s5, q5, NATOM / 16);
  k_bn_relu<float><<<cdiv((long long)NATOM * 128, 256), 256, 0, stream>>>(
      ZE2, edge_out, s5, q5, lift_g2, lift_b2, NATOM, 128);

  (void)in_sizes; (void)n_in; (void)out_size; (void)ws_size;
}
